// ResnetBBlock_67997922230599
// MI455X (gfx1250) — matrix operand order repaired and checked by reading, not compiled, not run on hardware
//
#include <hip/hip_runtime.h>

#define NPTS 100000
#define EDGES (NPTS * 16)
#define DIN 128
#define D2 64
#define DOUT 256
#define KP 15
#define SIGMA 0.04f
#define NEG 0.1f
#define BN_EPS 1e-5f

typedef __attribute__((ext_vector_type(16))) __bf16 v16bf;
typedef __attribute__((ext_vector_type(8)))  float v8f;
typedef __attribute__((ext_vector_type(4)))  int   v4i;

typedef __attribute__((address_space(1))) v4i* as1_v4i_ptr;
typedef __attribute__((address_space(3))) v4i* as3_v4i_ptr;

union BfTile { unsigned int u[8]; v16bf v; };

__device__ __forceinline__ unsigned short f32_to_bf16(float f) {
    unsigned int u = __float_as_uint(f);
    u += 0x7FFFu + ((u >> 16) & 1u);          // round-to-nearest-even
    return (unsigned short)(u >> 16);
}

// ---------------------------------------------------------------------------
// Zero fill
__global__ __launch_bounds__(256) void zero_f32(float* __restrict__ p, long long n) {
    long long i = (long long)blockIdx.x * 256 + threadIdx.x;
    if (i < n) p[i] = 0.0f;
}

// fp32 -> bf16 (same row-major layout)
__global__ __launch_bounds__(256) void cvt_bf16(const float* __restrict__ in,
                                                unsigned short* __restrict__ out,
                                                long long n) {
    long long i = (long long)blockIdx.x * 256 + threadIdx.x;
    if (i < n) out[i] = f32_to_bf16(in[i]);
}

// Weight convert + transpose: in fp32 [batch][R][C] -> out bf16 [batch][C][R]
__global__ __launch_bounds__(256) void wt_convert(const float* __restrict__ w,
                                                  unsigned short* __restrict__ wt,
                                                  int R, int C, int batch) {
    long long n = (long long)batch * R * C;
    long long i = (long long)blockIdx.x * 256 + threadIdx.x;
    if (i >= n) return;
    int b   = (int)(i / ((long long)R * C));
    int rem = (int)(i % ((long long)R * C));
    int r = rem / C;
    int c = rem % C;
    wt[(long long)b * R * C + (long long)c * R + r] = f32_to_bf16(w[i]);
}

// ---------------------------------------------------------------------------
// Tiled WMMA GEMM with async A-strip staging:
//   C[M,Nd] (+)= A[M,K]_bf16 @ B[K,Nd], B given transposed as Bt[Nd,K]_bf16.
// Block = 8 waves sharing one (Nd>=128) or two (Nd=64) 16-row A strips.
// A strip is copied global->LDS with GLOBAL_LOAD_ASYNC_TO_LDS_B128 (ASYNCcnt),
// then each wave owns a 16x16 output tile; A fragments come from LDS, B from
// the L2-resident transposed weights. Wave-uniform guards only => EXEC all-1s
// at every WMMA.
__global__ __launch_bounds__(256) void gemm_bf16_wmma(
        const unsigned short* __restrict__ A,
        const unsigned short* __restrict__ Bt,
        float* __restrict__ Cm,
        int M, int K, int Nd, int accumulate) {
    // Row stride 272B: every 16B async store aligned; (272/4)%64 == 4 banks
    // per row step -> the 16 row-lanes land on distinct banks.
    __shared__ unsigned short lA[2][16][136];

    const int ntn     = Nd >> 4;
    const int ntn_eff = (ntn < 8) ? ntn : 8;
    const int tmb     = 8 / ntn_eff;          // A strips per block (1 or 2)
    const int nblk_n  = ntn / ntn_eff;
    const int bm      = blockIdx.x / nblk_n;
    const int bn      = blockIdx.x - bm * nblk_n;
    const int tile_m0 = bm * tmb;

    // ---- cooperative async copy: A strip(s) -> LDS ----
    const int tid = threadIdx.x;
    const int cpr = K >> 3;                   // 16B chunks per 16-row
    const int nch = tmb * 16 * cpr;
    for (int ch = tid; ch < nch; ch += 256) {
        const int tm  = ch / (16 * cpr);
        const int rem = ch - tm * 16 * cpr;
        const int row = rem / cpr;
        const int cc  = rem - row * cpr;
        const unsigned short* src =
            A + (size_t)((tile_m0 + tm) * 16 + row) * K + cc * 8;
        unsigned short* dst = &lA[tm][row][cc * 8];
#if __has_builtin(__builtin_amdgcn_global_load_async_to_lds_b128)
        __builtin_amdgcn_global_load_async_to_lds_b128(
            (as1_v4i_ptr)(unsigned long long)src,
            (as3_v4i_ptr)(unsigned int)(unsigned long long)dst, 0, 0);
#else
        asm volatile("global_load_async_to_lds_b128 %0, %1, off"
                     :: "v"((unsigned)(unsigned long long)dst),
                        "v"((unsigned long long)src)
                     : "memory");
#endif
    }
#if __has_builtin(__builtin_amdgcn_s_wait_asynccnt)
    __builtin_amdgcn_s_wait_asynccnt(0);
#else
    asm volatile("s_wait_asynccnt 0x0" ::: "memory");
#endif
    __syncthreads();

    // ---- per-wave 16x16 tile ----
    const int wib    = tid >> 5;
    const int lane   = tid & 31;
    const int ltm    = wib / ntn_eff;
    const int tile_n = bn * ntn_eff + (wib - ltm * ntn_eff);
    const int tile_m = tile_m0 + ltm;
    const int ln     = lane & 15;
    const int half   = lane >> 4;

    const unsigned short* brow = Bt + (size_t)(tile_n * 16 + ln) * K;

    v8f c;
    if (accumulate) {
#pragma unroll
        for (int r = 0; r < 8; ++r)
            c[r] = Cm[(size_t)(tile_m * 16 + r + 8 * half) * Nd + tile_n * 16 + ln];
    } else {
#pragma unroll
        for (int r = 0; r < 8; ++r) c[r] = 0.0f;
    }

    for (int kk = 0; kk < K; kk += 32) {
        BfTile a, b;
#pragma unroll
        for (int j = 0; j < 8; ++j) {
            // A 16x32 bf16 layout: k = (j<4?0:16) + half*8 + 2*(j&3)
            const int ka = ((j & 4) << 2) + half * 8 + ((j & 3) << 1);
            a.u[j] = *(const unsigned int*)&lA[ltm][ln][kk + ka];
            // B 32x16 bf16 layout (from Bt rows): k = (j<4?0:16) + half*8 + 2*(j&3)
            const int kb = ((j & 4) << 2) + half * 8 + ((j & 3) << 1);
            b.u[j] = *(const unsigned int*)(brow + kk + kb);
        }
        c = __builtin_amdgcn_wmma_f32_16x16x32_bf16(
                false, a.v, false, b.v, (short)0, c, false, false);
    }

#pragma unroll
    for (int r = 0; r < 8; ++r)
        Cm[(size_t)(tile_m * 16 + r + 8 * half) * Nd + tile_n * 16 + ln] = c[r];
}

// ---------------------------------------------------------------------------
// Column sums / sums-of-squares over the point axis (training-mode BN stats).
// Block-local LDS reduction, one atomicAdd per column per block.
__global__ __launch_bounds__(256) void col_stats(const float* __restrict__ t,
                                                 int nrows, int C,
                                                 float* __restrict__ sums,
                                                 float* __restrict__ sumsq) {
    __shared__ float ls[256], lq[256];
    const int tid = threadIdx.x;
    const int c = tid % C;
    const int g = tid / C;
    const int ngroups = 256 / C;
    float s = 0.0f, q = 0.0f;
    for (int r = blockIdx.x * ngroups + g; r < nrows; r += gridDim.x * ngroups) {
        float v = t[(size_t)r * C + c];
        s += v; q += v * v;
    }
    ls[tid] = s; lq[tid] = q;
    __syncthreads();
    if (g == 0) {
        for (int gg = 1; gg < ngroups; ++gg) { s += ls[gg * C + c]; q += lq[gg * C + c]; }
        atomicAdd(&sums[c], s);
        atomicAdd(&sumsq[c], q);
    }
}

// BN apply (+ optional LeakyReLU, + optional bf16 mirror for next WMMA stage)
__global__ __launch_bounds__(256) void bn_apply(float* __restrict__ t,
                                                const float* __restrict__ sums,
                                                const float* __restrict__ sumsq,
                                                const float* __restrict__ g,
                                                const float* __restrict__ b,
                                                int nrows, int C, int do_leaky,
                                                unsigned short* __restrict__ out_bf16) {
    long long n = (long long)nrows * C;
    long long i = (long long)blockIdx.x * 256 + threadIdx.x;
    if (i >= n) return;
    int c = (int)(i % C);
    float invn = 1.0f / (float)nrows;
    float m = sums[c] * invn;
    float v = sumsq[c] * invn - m * m;            // biased variance, like torch/jnp
    float sc = g[c] * rsqrtf(v + BN_EPS);
    float y = (t[i] - m) * sc + b[c];
    if (do_leaky) y = (y >= 0.0f) ? y : NEG * y;
    t[i] = y;
    if (out_bf16) out_bf16[i] = f32_to_bf16(y);
}

// ---------------------------------------------------------------------------
// KPConv scatter for one kernel point k:
//   agg[q, :] += max(0, 1 - |rel - kp_k| / sigma) * h[r, :]
// 4 edges per block; 64 lanes per edge (one per channel). Influence computed
// redundantly per lane (broadcast loads); edges outside the influence radius
// (the common case here) skip the 256B gather + 256B atomic entirely.
__global__ __launch_bounds__(256) void kp_scatter(float* __restrict__ agg,
                                                  const float* __restrict__ h,
                                                  const float* __restrict__ pos,
                                                  const int* __restrict__ e_ref,
                                                  const int* __restrict__ e_query,
                                                  const float* __restrict__ kpts,
                                                  int k) {
    const int tid  = threadIdx.x;
    const int c    = tid & 63;
    const long long e = (long long)blockIdx.x * 4 + (tid >> 6);
    if (e >= EDGES) return;
    const int r = e_ref[e];
    const int q = e_query[e];
    const float dx = pos[r * 4 + 1] - pos[q * 4 + 1] - kpts[k * 3 + 0];
    const float dy = pos[r * 4 + 2] - pos[q * 4 + 2] - kpts[k * 3 + 1];
    const float dz = pos[r * 4 + 3] - pos[q * 4 + 3] - kpts[k * 3 + 2];
    const float w = 1.0f - sqrtf(dx * dx + dy * dy + dz * dz) * (1.0f / SIGMA);
    if (w <= 0.0f) return;
    atomicAdd(&agg[(size_t)q * D2 + c], w * h[(size_t)r * D2 + c]);
}

// ---------------------------------------------------------------------------
// Final: out = leaky(BN(y2)) + BN(skip), in place on d_out
__global__ __launch_bounds__(256) void final_fuse(float* __restrict__ out,
                                                  const float* __restrict__ skip,
                                                  const float* __restrict__ s2,
                                                  const float* __restrict__ q2,
                                                  const float* __restrict__ g2,
                                                  const float* __restrict__ b2,
                                                  const float* __restrict__ ssc,
                                                  const float* __restrict__ qsc,
                                                  const float* __restrict__ gsc,
                                                  const float* __restrict__ bsc) {
    long long n = (long long)NPTS * DOUT;
    long long i = (long long)blockIdx.x * 256 + threadIdx.x;
    if (i >= n) return;
    int c = (int)(i % DOUT);
    const float invn = 1.0f / (float)NPTS;
    float m2 = s2[c] * invn;
    float v2 = q2[c] * invn - m2 * m2;
    float y  = (out[i] - m2) * (g2[c] * rsqrtf(v2 + BN_EPS)) + b2[c];
    y = (y >= 0.0f) ? y : NEG * y;
    float ms = ssc[c] * invn;
    float vs = qsc[c] * invn - ms * ms;
    float sk = (skip[i] - ms) * (gsc[c] * rsqrtf(vs + BN_EPS)) + bsc[c];
    out[i] = y + sk;
}

// ---------------------------------------------------------------------------
extern "C" void kernel_launch(void* const* d_in, const int* in_sizes, int n_in,
                              void* d_out, int out_size, void* d_ws, size_t ws_size,
                              hipStream_t stream) {
    const float* pos    = (const float*)d_in[0];
    const float* x      = (const float*)d_in[1];
    const int*   e_ref  = (const int*)  d_in[2];
    const int*   e_query= (const int*)  d_in[3];
    const float* W1     = (const float*)d_in[4];
    const float* g1     = (const float*)d_in[5];
    const float* b1     = (const float*)d_in[6];
    const float* kpts   = (const float*)d_in[7];
    const float* W_kp   = (const float*)d_in[8];
    const float* gkp    = (const float*)d_in[9];
    const float* bkp    = (const float*)d_in[10];
    const float* W2     = (const float*)d_in[11];
    const float* g2     = (const float*)d_in[12];
    const float* b2     = (const float*)d_in[13];
    const float* Wsc    = (const float*)d_in[14];
    const float* gsc    = (const float*)d_in[15];
    const float* bsc    = (const float*)d_in[16];
    float* out = (float*)d_out;

    // ---- workspace carve-up (256B aligned regions) ----
    size_t off = 0;
    char* base = (char*)d_ws;
    auto take = [&](size_t bytes) -> char* {
        char* p = base + off;
        off += (bytes + 255) & ~(size_t)255;
        return p;
    };
    unsigned short* xb   = (unsigned short*)take((size_t)NPTS * DIN * 2);  // x bf16
    float*          h    = (float*)         take((size_t)NPTS * D2  * 4);  // t1 -> h (in place)
    float*          agg  = (float*)         take((size_t)NPTS * D2  * 4);  // per-k segment sums
    unsigned short* aggb = (unsigned short*)take((size_t)NPTS * D2  * 2);
    float*          y    = (float*)         take((size_t)NPTS * D2  * 4);  // KPConv accumulator
    unsigned short* yb   = (unsigned short*)take((size_t)NPTS * D2  * 2);
    float*          skip = (float*)         take((size_t)NPTS * DOUT* 4);
    unsigned short* w1t  = (unsigned short*)take((size_t)D2   * DIN * 2);
    unsigned short* w2t  = (unsigned short*)take((size_t)DOUT * D2  * 2);
    unsigned short* wsct = (unsigned short*)take((size_t)DOUT * DIN * 2);
    unsigned short* wkpt = (unsigned short*)take((size_t)KP * D2 * D2 * 2);
    float* stats = (float*)take(1280 * 4);
    float* s1  = stats;        float* q1  = stats + 64;
    float* skp = stats + 128;  float* qkp = stats + 192;
    float* s2  = stats + 256;  float* q2  = stats + 512;
    float* ssc = stats + 768;  float* qsc = stats + 1024;

    auto blocks = [](long long n) { return (unsigned)((n + 255) / 256); };
    // gemm grid: (#m-strips / strips-per-block) * n-block-splits
    auto gemm_blocks = [](int M, int Nd) {
        int ntn = Nd >> 4;
        int ntn_eff = (ntn < 8) ? ntn : 8;
        int tmb = 8 / ntn_eff;
        int nblk_n = ntn / ntn_eff;
        return (unsigned)(((M >> 4) / tmb) * nblk_n);
    };

    // 0) zero accumulators (every call: deterministic)
    zero_f32<<<blocks(1280), 256, 0, stream>>>(stats, 1280);
    zero_f32<<<blocks((long long)NPTS * D2), 256, 0, stream>>>(y, (long long)NPTS * D2);

    // 1) precision staging: x and weights -> bf16 (weights transposed to [n][k])
    cvt_bf16<<<blocks((long long)NPTS * DIN), 256, 0, stream>>>(x, xb, (long long)NPTS * DIN);
    wt_convert<<<blocks((long long)DIN * D2),        256, 0, stream>>>(W1,   w1t,  DIN, D2,  1);
    wt_convert<<<blocks((long long)D2 * DOUT),       256, 0, stream>>>(W2,   w2t,  D2,  DOUT, 1);
    wt_convert<<<blocks((long long)DIN * DOUT),      256, 0, stream>>>(Wsc,  wsct, DIN, DOUT, 1);
    wt_convert<<<blocks((long long)KP * D2 * D2),    256, 0, stream>>>(W_kp, wkpt, D2,  D2,  KP);

    // 2) bottleneck unary_1: t1 = x @ W1 ; BN stats; BN+leaky in place -> h
    gemm_bf16_wmma<<<gemm_blocks(NPTS, D2), 256, 0, stream>>>(xb, w1t, h, NPTS, DIN, D2, 0);
    col_stats<<<512, 256, 0, stream>>>(h, NPTS, D2, s1, q1);
    bn_apply<<<blocks((long long)NPTS * D2), 256, 0, stream>>>(h, s1, q1, g1, b1, NPTS, D2, 1, (unsigned short*)nullptr);

    // 3) KPConv: per kernel point — zero, scatter, cvt, accumulated WMMA GEMM
    for (int k = 0; k < KP; ++k) {
        zero_f32<<<blocks((long long)NPTS * D2), 256, 0, stream>>>(agg, (long long)NPTS * D2);
        kp_scatter<<<EDGES / 4, 256, 0, stream>>>(agg, h, pos, e_ref, e_query, kpts, k);
        cvt_bf16<<<blocks((long long)NPTS * D2), 256, 0, stream>>>(agg, aggb, (long long)NPTS * D2);
        gemm_bf16_wmma<<<gemm_blocks(NPTS, D2), 256, 0, stream>>>(
            aggb, wkpt + (size_t)k * D2 * D2, y, NPTS, D2, D2, 1);
    }

    // 4) SimpleBlock post BN+leaky (also emit bf16 for unary_2 GEMM)
    col_stats<<<512, 256, 0, stream>>>(y, NPTS, D2, skp, qkp);
    bn_apply<<<blocks((long long)NPTS * D2), 256, 0, stream>>>(y, skp, qkp, gkp, bkp, NPTS, D2, 1, yb);

    // 5) unary_2 pre-BN: y2 = y @ W2  (straight into d_out, fp32)
    gemm_bf16_wmma<<<gemm_blocks(NPTS, DOUT), 256, 0, stream>>>(yb, w2t, out, NPTS, D2, DOUT, 0);
    col_stats<<<512, 256, 0, stream>>>(out, NPTS, DOUT, s2, q2);

    // 6) shortcut pre-BN: skip = x @ Wsc
    gemm_bf16_wmma<<<gemm_blocks(NPTS, DOUT), 256, 0, stream>>>(xb, wsct, skip, NPTS, DIN, DOUT, 0);
    col_stats<<<512, 256, 0, stream>>>(skip, NPTS, DOUT, ssc, qsc);

    // 7) out = leaky(BN(y2)) + BN(skip)
    final_fuse<<<blocks((long long)NPTS * DOUT), 256, 0, stream>>>(
        out, skip, s2, q2, g2, b2, ssc, qsc, gsc, bsc);
}
